// Model_41515153883377
// MI455X (gfx1250) — compile-verified, matrix operand order checked by reading, not run on hardware
//
#include <hip/hip_runtime.h>
#include <hip/hip_bf16.h>

// ---------------------------------------------------------------------------
// CDNA5 (gfx1250) implementation: all matmuls via v_wmma_f32_16x16x32_f16.
// wave32; blocks of 128 threads = 4 waves; each wave owns one 16-row M tile.
// v2: register-hoisted fragments, branch-free TRANS gate math, LDS-staged
//     per-row aux indices, global_prefetch_b8 hints.
// ---------------------------------------------------------------------------

typedef __attribute__((ext_vector_type(16))) _Float16 v16h;
typedef __attribute__((ext_vector_type(8)))  float    v8f;

#define B_     256
#define CITY_  184
#define GROUP_ 16
#define T_     24
#define FIN_   8
#define XEM_   64
#define DATE_  8
#define LOCEM_ 16
#define EDGEH_ 32
#define GNNH_  64
#define E_     2000
#define P_     240            // GROUP*(GROUP-1)
#define NSEQ_  (B_*CITY_)     // 47104
#define NGRP_  (B_*GROUP_)    // 4096
#define NGE_   (B_*P_)        // 61440
#define NCE_   (B_*E_)        // 512000

// ---- WMMA fragment helpers (layouts per ISA 7.12.2, 16-bit A 16x32 / B 32x16)
__device__ __forceinline__ v8f wmma_f16(v8f acc, v16h a, v16h b) {
  return __builtin_amdgcn_wmma_f32_16x16x32_f16(false, a, false, b,
                                                (short)0, acc, false, false);
}

// A fragment: 16x32 tile, rows m = lane&15, halves map K = 16*(h>>3)+8*(lane>>4)+(h&7)
__device__ __forceinline__ v16h frag_a(const _Float16* Asm, int ld, int lane, int k0) {
  const int m  = lane & 15;
  const int kb = k0 + ((lane >> 4) << 3);
  const _Float16* p = Asm + m * ld;
  v16h f;
#pragma unroll
  for (int h = 0; h < 16; ++h)
    f[h] = p[kb + ((h >> 3) << 4) + (h & 7)];
  return f;
}

// B fragment: 32x16 tile; weights stored row-major W[n][k] (so B[k][n] = W[n][k])
__device__ __forceinline__ v16h frag_b(const _Float16* Wsm, int ld, int lane,
                                       int k0, int n0) {
  const int n  = n0 + (lane & 15);
  const int kb = k0 + ((lane >> 4) << 3);
  const _Float16* p = Wsm + n * ld;
  v16h f;
#pragma unroll
  for (int h = 0; h < 16; ++h)
    f[h] = p[kb + ((h >> 3) << 4) + (h & 7)];
  return f;
}

// branch-free TRANS-unit sigmoid/tanh (v_exp_f32 + v_rcp_f32; co-execute w/ WMMA)
__device__ __forceinline__ float sigf(float v) {
  return __builtin_amdgcn_rcpf(1.f + __expf(-v));
}
__device__ __forceinline__ float tanhf_fast(float v) {
  return 1.f - 2.f * __builtin_amdgcn_rcpf(1.f + __expf(2.f * v));
}

__device__ __forceinline__ void pairIJ(int p, int& i, int& j) {
  i = p / 15;
  int q = p - i * 15;
  j = q + (q >= i ? 1 : 0);
}

// ---------------------------------------------------------------------------
// 1) LSTM: 16 sequences/block, 24 timesteps, gates = A[16x96] @ W[96x256]
//    A row = [x_t(8) | h(64) | pad(24)].  W fragments hoisted to registers:
//    steady-state loop = 6 ds_load_b128 (A frags) + 12 v_wmma + gate TRANS ops.
// ---------------------------------------------------------------------------
__global__ __launch_bounds__(128) void lstm_k(const float* __restrict__ x,
                                              const float* __restrict__ Wih,
                                              const float* __restrict__ Whh,
                                              const float* __restrict__ bih,
                                              const float* __restrict__ bhh,
                                              _Float16* __restrict__ xe16) {
  __shared__ _Float16 sW[256 * 96];   // 48 KB
  __shared__ float    sB[256];
  __shared__ _Float16 sA[16 * 96];    // 3 KB
  const int tid = threadIdx.x, lane = tid & 31, wave = tid >> 5;

  // prefetch weights into cache while LDS fills
  __builtin_prefetch(Whh + (size_t)tid * 128, 0, 0);

  for (int e = tid; e < 256 * 96; e += 128) {
    const int n = e / 96, k = e - n * 96;
    float w = 0.f;
    if (k < 8)       w = Wih[n * 8 + k];
    else if (k < 72) w = Whh[n * 64 + (k - 8)];
    sW[e] = (_Float16)w;
  }
  for (int n = tid; n < 256; n += 128) sB[n] = bih[n] + bhh[n];
  for (int e = tid; e < 16 * 96; e += 128) sA[e] = (_Float16)0.f;
  __syncthreads();

  // hoist all 12 B fragments (gate x k-step) into registers: 96 VGPRs
  v16h bfr[12];
#pragma unroll
  for (int ks = 0; ks < 3; ++ks)
#pragma unroll
    for (int g = 0; g < 4; ++g)
      bfr[ks * 4 + g] = frag_b(sW, 96, lane, ks * 32, g * 64 + wave * 16);

  const int seq0 = blockIdx.x * 16;
  const int j    = wave * 16 + (lane & 15);   // hidden unit
  const int mb   = (lane >> 4) << 3;          // row base within tile
  const float bI = sB[j], bF = sB[64 + j], bG = sB[128 + j], bO = sB[192 + j];
  const int xm = tid >> 3, xf = tid & 7;      // this thread's x element
  float c[8];
#pragma unroll
  for (int r = 0; r < 8; ++r) c[r] = 0.f;

  for (int t = 0; t < T_; ++t) {
    // write x_t rows (16 x 8) into A; prefetch next timestep's slice
    sA[xm * 96 + xf] = (_Float16)x[((size_t)(seq0 + xm) * T_ + t) * FIN_ + xf];
    if (t + 1 < T_)
      __builtin_prefetch(&x[((size_t)(seq0 + xm) * T_ + t + 1) * FIN_ + xf], 0, 0);
    __syncthreads();
    v8f aI = {}, aF = {}, aG = {}, aO = {};
#pragma unroll
    for (int ks = 0; ks < 3; ++ks) {
      const v16h av = frag_a(sA, 96, lane, ks * 32);
      aI = wmma_f16(aI, av, bfr[ks * 4 + 0]);
      aF = wmma_f16(aF, av, bfr[ks * 4 + 1]);
      aG = wmma_f16(aG, av, bfr[ks * 4 + 2]);
      aO = wmma_f16(aO, av, bfr[ks * 4 + 3]);
    }
    __syncthreads();
#pragma unroll
    for (int r = 0; r < 8; ++r) {
      const float gi = sigf(aI[r] + bI);
      const float gf = sigf(aF[r] + bF);
      const float gg = tanhf_fast(aG[r] + bG);
      const float go = sigf(aO[r] + bO);
      c[r] = gf * c[r] + gi * gg;
      const float hv = go * tanhf_fast(c[r]);
      sA[(mb + r) * 96 + 8 + j] = (_Float16)hv;       // h feeds next step
      if (t == T_ - 1)
        xe16[(size_t)(seq0 + mb + r) * XEM_ + j] = (_Float16)hv;
    }
    __syncthreads();
  }
}

// ---------------------------------------------------------------------------
// 2) Generic WMMA GEMM (64 rows per block) with per-layer gather/emit policy.
//    Per-row aux values (packed src/dst node ids) staged once in LDS so the
//    per-element gather does no integer div/mod and no index reloads.
// ---------------------------------------------------------------------------
struct GArgs {
  const float*    W;    const float* bias;
  const _Float16* H0;   const _Float16* H1;
  const float*    F0;   const float*    F1;
  const int*      I0;
  float*          OutF; _Float16*       OutH;
  int             K;
};

template <class Pol>
__global__ __launch_bounds__(128) void gemm_k(GArgs a) {
  constexpr int KP = Pol::KPAD, N = Pol::N;
  constexpr int KS = KP / 32,   NT = N / 16;
  __shared__ _Float16 sW[N * KP];
  __shared__ float    sB[N];
  __shared__ _Float16 sA[64 * KP];
  __shared__ int      sAuxA[64];
  __shared__ int      sAuxB[64];
  const int tid = threadIdx.x, lane = tid & 31, wave = tid >> 5;
  const int row0 = blockIdx.x * 64;

  __builtin_prefetch(a.W + (size_t)tid * 16, 0, 0);

  for (int e = tid; e < N * KP; e += 128) {
    const int n = e / KP, k = e - n * KP;
    sW[e] = (k < a.K) ? (_Float16)a.W[n * a.K + k] : (_Float16)0.f;
  }
  for (int n = tid; n < N; n += 128) sB[n] = a.bias[n];
  if (tid < 64) {
    sAuxA[tid] = Pol::auxA(a, row0 + tid);
    sAuxB[tid] = Pol::auxB(a, row0 + tid);
  }
  __syncthreads();

  for (int e = tid; e < 64 * KP; e += 128) {
    const int r = e / KP, k = e - r * KP;
    sA[e] = Pol::gather(a, row0 + r, k, sAuxA[r]);
  }
  __syncthreads();

  const _Float16* Ar = sA + wave * 16 * KP;
  v16h af[KS];
#pragma unroll
  for (int ks = 0; ks < KS; ++ks) af[ks] = frag_a(Ar, KP, lane, ks * 32);

  for (int nt = 0; nt < NT; ++nt) {
    v8f acc = {};
#pragma unroll
    for (int ks = 0; ks < KS; ++ks)
      acc = wmma_f16(acc, af[ks], frag_b(sW, KP, lane, ks * 32, nt * 16));
    const int n   = nt * 16 + (lane & 15);
    const int r0  = wave * 16 + ((lane >> 4) << 3);   // tile-local row base
    const float bb = sB[n];
#pragma unroll
    for (int r = 0; r < 8; ++r) {
      float v = acc[r] + bb;
      v = v > 0.f ? v : 0.f;                          // relu
      Pol::emit(a, row0 + r0 + r, n, v, sAuxB[r0 + r]);
    }
  }
}

// ---- helpers for aux packing (group edges): aux = (b<<8)|(i<<4)|j ----
__device__ __forceinline__ int packGroupAux(int row) {
  const int b = row / P_, p = row - b * P_;
  int i, j; pairIJ(p, i, j);
  return (b << 8) | (i << 4) | j;
}

// ---- policies ----
// edge_inf: rows=B*P, A=[gi(80)|gj(80)|ue(24)] -> 32, store f16
struct PolEdgeInf {
  static constexpr int KPAD = 192, N = 32;
  static __device__ int auxA(const GArgs&, int row) { return packGroupAux(row); }
  static __device__ int auxB(const GArgs&, int)     { return 0; }
  static __device__ _Float16 gather(const GArgs& a, int, int k, int s) {
    const int b = s >> 8;
    if (k < 80)  return a.H0[(size_t)((b << 4) + ((s >> 4) & 15)) * 80 + k];
    if (k < 160) return a.H0[(size_t)((b << 4) + (s & 15)) * 80 + (k - 80)];
    if (k < 184) return a.H1[(size_t)b * 24 + (k - 160)];
    return (_Float16)0.f;
  }
  static __device__ void emit(const GArgs& a, int row, int n, float v, int) {
    a.OutH[(size_t)row * N + n] = (_Float16)v;
  }
};
// group GNN L1 msg: A=[gx[src](80)|gew(32)] -> 64, scatter-add to dst node
struct PolGMsg1 {
  static constexpr int KPAD = 128, N = 64;
  static __device__ int auxA(const GArgs&, int row) { return packGroupAux(row); }
  static __device__ int auxB(const GArgs&, int row) {
    const int s = packGroupAux(row);
    return ((s >> 8) << 4) + (s & 15);                // dst node = b*16 + j
  }
  static __device__ _Float16 gather(const GArgs& a, int row, int k, int s) {
    if (k < 80)  return a.H0[(size_t)(((s >> 8) << 4) + ((s >> 4) & 15)) * 80 + k];
    if (k < 112) return a.H1[(size_t)row * EDGEH_ + (k - 80)];
    return (_Float16)0.f;
  }
  static __device__ void emit(const GArgs& a, int, int n, float v, int d) {
    atomicAdd(&a.OutF[(size_t)d * GNNH_ + n], v);
  }
};
// group GNN L1 node: A=[gx(80)|agg/15(64)] -> 64
struct PolGNode1 {
  static constexpr int KPAD = 160, N = 64;
  static __device__ int auxA(const GArgs&, int) { return 0; }
  static __device__ int auxB(const GArgs&, int) { return 0; }
  static __device__ _Float16 gather(const GArgs& a, int row, int k, int) {
    if (k < 80)  return a.H0[(size_t)row * 80 + k];
    if (k < 144) return (_Float16)(a.F0[(size_t)row * GNNH_ + (k - 80)] * (1.f / 15.f));
    return (_Float16)0.f;
  }
  static __device__ void emit(const GArgs& a, int row, int n, float v, int) {
    a.OutH[(size_t)row * N + n] = (_Float16)v;
  }
};
// group GNN L2 msg: A=[gx1(64)|gew(32)] -> 64, scatter-add
struct PolGMsg2 {
  static constexpr int KPAD = 96, N = 64;
  static __device__ int auxA(const GArgs&, int row) { return packGroupAux(row); }
  static __device__ int auxB(const GArgs&, int row) {
    const int s = packGroupAux(row);
    return ((s >> 8) << 4) + (s & 15);
  }
  static __device__ _Float16 gather(const GArgs& a, int row, int k, int s) {
    if (k < 64) return a.H0[(size_t)(((s >> 8) << 4) + ((s >> 4) & 15)) * GNNH_ + k];
    if (k < 96) return a.H1[(size_t)row * EDGEH_ + (k - 64)];
    return (_Float16)0.f;
  }
  static __device__ void emit(const GArgs& a, int, int n, float v, int d) {
    atomicAdd(&a.OutF[(size_t)d * GNNH_ + n], v);
  }
};
// group GNN L2 node: A=[gx1(64)|agg/15(64)] -> 64
struct PolGNode2 {
  static constexpr int KPAD = 128, N = 64;
  static __device__ int auxA(const GArgs&, int) { return 0; }
  static __device__ int auxB(const GArgs&, int) { return 0; }
  static __device__ _Float16 gather(const GArgs& a, int row, int k, int) {
    if (k < 64)  return a.H0[(size_t)row * GNNH_ + k];
    if (k < 128) return (_Float16)(a.F0[(size_t)row * GNNH_ + (k - 64)] * (1.f / 15.f));
    return (_Float16)0.f;
  }
  static __device__ void emit(const GArgs& a, int row, int n, float v, int) {
    a.OutH[(size_t)row * N + n] = (_Float16)v;
  }
};
// global GNN L1 msg: A=[new_x[src](128)|edge_w(1)] -> 64, scatter-add
struct PolCMsg1 {
  static constexpr int KPAD = 160, N = 64;
  static __device__ int auxA(const GArgs& a, int row) {
    const int b = row / E_, e = row - b * E_;
    return b * CITY_ + a.I0[(size_t)b * 2 * E_ + e];
  }
  static __device__ int auxB(const GArgs& a, int row) {
    const int b = row / E_, e = row - b * E_;
    return b * CITY_ + a.I0[(size_t)b * 2 * E_ + E_ + e];
  }
  static __device__ _Float16 gather(const GArgs& a, int row, int k, int s) {
    if (k < 128)  return a.H0[(size_t)s * 128 + k];
    if (k == 128) return (_Float16)a.F0[row];
    return (_Float16)0.f;
  }
  static __device__ void emit(const GArgs& a, int, int n, float v, int d) {
    atomicAdd(&a.OutF[(size_t)d * GNNH_ + n], v);
  }
};
// global GNN L1 node: A=[new_x(128)|agg/cnt(64)] -> 64
struct PolCNode1 {
  static constexpr int KPAD = 192, N = 64;
  static __device__ int auxA(const GArgs&, int) { return 0; }
  static __device__ int auxB(const GArgs&, int) { return 0; }
  static __device__ _Float16 gather(const GArgs& a, int row, int k, int) {
    if (k < 128) return a.H0[(size_t)row * 128 + k];
    return (_Float16)(a.F0[(size_t)row * GNNH_ + (k - 128)] / fmaxf(a.F1[row], 1.f));
  }
  static __device__ void emit(const GArgs& a, int row, int n, float v, int) {
    a.OutH[(size_t)row * N + n] = (_Float16)v;
  }
};
// global GNN L2 msg: A=[h1[src](64)|edge_w(1)] -> 64, scatter-add
struct PolCMsg2 {
  static constexpr int KPAD = 96, N = 64;
  static __device__ int auxA(const GArgs& a, int row) {
    const int b = row / E_, e = row - b * E_;
    return b * CITY_ + a.I0[(size_t)b * 2 * E_ + e];
  }
  static __device__ int auxB(const GArgs& a, int row) {
    const int b = row / E_, e = row - b * E_;
    return b * CITY_ + a.I0[(size_t)b * 2 * E_ + E_ + e];
  }
  static __device__ _Float16 gather(const GArgs& a, int row, int k, int s) {
    if (k < 64)  return a.H0[(size_t)s * GNNH_ + k];
    if (k == 64) return (_Float16)a.F0[row];
    return (_Float16)0.f;
  }
  static __device__ void emit(const GArgs& a, int, int n, float v, int d) {
    atomicAdd(&a.OutF[(size_t)d * GNNH_ + n], v);
  }
};
// global GNN L2 node: A=[h1(64)|agg/cnt(64)] -> 64
struct PolCNode2 {
  static constexpr int KPAD = 128, N = 64;
  static __device__ int auxA(const GArgs&, int) { return 0; }
  static __device__ int auxB(const GArgs&, int) { return 0; }
  static __device__ _Float16 gather(const GArgs& a, int row, int k, int) {
    if (k < 64) return a.H0[(size_t)row * GNNH_ + k];
    return (_Float16)(a.F0[(size_t)row * GNNH_ + (k - 64)] / fmaxf(a.F1[row], 1.f));
  }
  static __device__ void emit(const GArgs& a, int row, int n, float v, int) {
    a.OutH[(size_t)row * N + n] = (_Float16)v;
  }
};
// pred layers
struct PolPred0 {
  static constexpr int KPAD = 64, N = 128;
  static __device__ int auxA(const GArgs&, int) { return 0; }
  static __device__ int auxB(const GArgs&, int) { return 0; }
  static __device__ _Float16 gather(const GArgs& a, int row, int k, int) {
    return a.H0[(size_t)row * 64 + k];
  }
  static __device__ void emit(const GArgs& a, int row, int n, float v, int) {
    a.OutH[(size_t)row * N + n] = (_Float16)v;
  }
};
struct PolPred1 {
  static constexpr int KPAD = 128, N = 128;
  static __device__ int auxA(const GArgs&, int) { return 0; }
  static __device__ int auxB(const GArgs&, int) { return 0; }
  static __device__ _Float16 gather(const GArgs& a, int row, int k, int) {
    return a.H0[(size_t)row * 128 + k];
  }
  static __device__ void emit(const GArgs& a, int row, int n, float v, int) {
    a.OutH[(size_t)row * N + n] = (_Float16)v;
  }
};

// ---------------------------------------------------------------------------
// 3) small VALU kernels
// ---------------------------------------------------------------------------
__global__ void soft_k(const float* __restrict__ wg, float* __restrict__ ws) {
  const int c = blockIdx.x * blockDim.x + threadIdx.x;
  if (c >= CITY_) return;
  float mx = -1e30f;
  for (int g = 0; g < GROUP_; ++g) mx = fmaxf(mx, wg[c * GROUP_ + g]);
  float e[GROUP_], s = 0.f;
  for (int g = 0; g < GROUP_; ++g) { e[g] = __expf(wg[c * GROUP_ + g] - mx); s += e[g]; }
  const float rs = __builtin_amdgcn_rcpf(s);
  for (int g = 0; g < GROUP_; ++g) ws[c * GROUP_ + g] = e[g] * rs;
}

__global__ void uem_k(const int* __restrict__ u, const float* __restrict__ e1,
                      const float* __restrict__ e2, const float* __restrict__ e3,
                      _Float16* __restrict__ ue) {
  const int b = blockIdx.x, t = threadIdx.x;
  if (t >= 24) return;
  const int slot = t >> 3, q = t & 7;
  const float* tab = (slot == 0) ? e1 : (slot == 1) ? e2 : e3;
  ue[b * 24 + t] = (_Float16)tab[u[b * 3 + slot] * DATE_ + q];
}

// g_x[b][g][f] = sum_c wsoft[c][g] * [xe | loc@Wloc^T+b][b][c][f]   (f < 80)
__global__ void gx_k(const _Float16* __restrict__ xe16, const float* __restrict__ loc,
                     const float* __restrict__ Wloc, const float* __restrict__ bloc,
                     const float* __restrict__ wsoft, _Float16* __restrict__ gx) {
  const int bg = blockIdx.x;
  const int b = bg >> 4, g = bg & 15;
  const int f = threadIdx.x;
  if (f >= 80) return;
  float acc = 0.f;
  for (int c = 0; c < CITY_; ++c) {
    const float w = wsoft[c * GROUP_ + g];
    float v;
    if (f < 64) {
      v = (float)xe16[(size_t)(b * CITY_ + c) * XEM_ + f];
    } else {
      const int q = f - 64;
      const float* lp = loc + (size_t)(b * CITY_ + c) * 2;
      v = lp[0] * Wloc[q * 2] + lp[1] * Wloc[q * 2 + 1] + bloc[q];
    }
    acc += w * v;
  }
  gx[(size_t)bg * 80 + f] = (_Float16)acc;
}

// new_x[b][c] = [xe(64) | sum_g wsoft[c][g]*gx2[b][g](64)]
__global__ void newx_k(const _Float16* __restrict__ xe16, const _Float16* __restrict__ gx2,
                       const float* __restrict__ wsoft, _Float16* __restrict__ nx) {
  const int bc = blockIdx.x;
  const int b = bc / CITY_, c = bc - b * CITY_;
  const int f = threadIdx.x;   // 64
  nx[(size_t)bc * 128 + f] = xe16[(size_t)bc * XEM_ + f];
  float acc = 0.f;
  for (int g = 0; g < GROUP_; ++g)
    acc += wsoft[c * GROUP_ + g] * (float)gx2[(size_t)(b * GROUP_ + g) * GNNH_ + f];
  nx[(size_t)bc * 128 + 64 + f] = (_Float16)acc;
}

__global__ void count_k(const int* __restrict__ ei, float* __restrict__ cnt) {
  const int ee = blockIdx.x * blockDim.x + threadIdx.x;
  if (ee >= NCE_) return;
  const int b = ee / E_, e = ee - b * E_;
  const int d = ei[(size_t)b * 2 * E_ + E_ + e];
  atomicAdd(&cnt[b * CITY_ + d], 1.0f);
}

__global__ void pred2_k(const _Float16* __restrict__ q2, const float* __restrict__ W,
                        const float* __restrict__ bias, float* __restrict__ out) {
  const int n = blockIdx.x * blockDim.x + threadIdx.x;
  if (n >= NSEQ_) return;
  float acc = bias[0];
  for (int k = 0; k < 128; ++k) acc += (float)q2[(size_t)n * 128 + k] * W[k];
  out[n] = fmaxf(acc, 0.f);
}

// ---------------------------------------------------------------------------
// host launcher
// ---------------------------------------------------------------------------
extern "C" void kernel_launch(void* const* d_in, const int* in_sizes, int n_in,
                              void* d_out, int out_size, void* d_ws, size_t ws_size,
                              hipStream_t stream) {
  // Inputs: x,u,edge_index,edge_w,loc then params flattened in sorted pytree order:
  // edge_inf(W,b), emb1, emb2, emb3, global_gnn[0..1](mlp1 W,b; mlp2 W,b),
  // group_gnn[0..1](...), loc(W,b), lstm(Whh,Wih,bhh,bih), pred[0..2](W,b), w_group
  const float* x    = (const float*)d_in[0];
  const int*   u    = (const int*)  d_in[1];
  const int*   ei   = (const int*)  d_in[2];
  const float* ew   = (const float*)d_in[3];
  const float* loc  = (const float*)d_in[4];
  const float* Wedge = (const float*)d_in[5];
  const float* bedge = (const float*)d_in[6];
  const float* emb1 = (const float*)d_in[7];
  const float* emb2 = (const float*)d_in[8];
  const float* emb3 = (const float*)d_in[9];
  const float* Wc1m1 = (const float*)d_in[10]; const float* bc1m1 = (const float*)d_in[11];
  const float* Wc1m2 = (const float*)d_in[12]; const float* bc1m2 = (const float*)d_in[13];
  const float* Wc2m1 = (const float*)d_in[14]; const float* bc2m1 = (const float*)d_in[15];
  const float* Wc2m2 = (const float*)d_in[16]; const float* bc2m2 = (const float*)d_in[17];
  const float* Wg1m1 = (const float*)d_in[18]; const float* bg1m1 = (const float*)d_in[19];
  const float* Wg1m2 = (const float*)d_in[20]; const float* bg1m2 = (const float*)d_in[21];
  const float* Wg2m1 = (const float*)d_in[22]; const float* bg2m1 = (const float*)d_in[23];
  const float* Wg2m2 = (const float*)d_in[24]; const float* bg2m2 = (const float*)d_in[25];
  const float* Wloc = (const float*)d_in[26]; const float* bloc = (const float*)d_in[27];
  const float* Whh  = (const float*)d_in[28]; const float* Wih  = (const float*)d_in[29];
  const float* bhh  = (const float*)d_in[30]; const float* bih  = (const float*)d_in[31];
  const float* Wp0 = (const float*)d_in[32]; const float* bp0 = (const float*)d_in[33];
  const float* Wp1 = (const float*)d_in[34]; const float* bp1 = (const float*)d_in[35];
  const float* Wp2 = (const float*)d_in[36]; const float* bp2 = (const float*)d_in[37];
  const float* wgrp = (const float*)d_in[38];
  float* out = (float*)d_out;

  // workspace carve-up
  char* ws = (char*)d_ws; size_t off = 0;
  auto alloc = [&](size_t bytes) -> void* {
    void* p = ws + off; off = (off + bytes + 255) & ~(size_t)255; return p;
  };
  _Float16* xe16  = (_Float16*)alloc((size_t)NSEQ_ * XEM_ * 2);
  float*    wsoft = (float*)   alloc((size_t)CITY_ * GROUP_ * 4);
  _Float16* ue16  = (_Float16*)alloc((size_t)B_ * 24 * 2);
  _Float16* gx    = (_Float16*)alloc((size_t)NGRP_ * 80 * 2);
  _Float16* gew   = (_Float16*)alloc((size_t)NGE_ * EDGEH_ * 2);
  float*    aggG  = (float*)   alloc((size_t)NGRP_ * GNNH_ * 4);
  _Float16* gx1   = (_Float16*)alloc((size_t)NGRP_ * GNNH_ * 2);
  _Float16* gx2   = (_Float16*)alloc((size_t)NGRP_ * GNNH_ * 2);
  _Float16* nx    = (_Float16*)alloc((size_t)NSEQ_ * 128 * 2);
  float*    aggC  = (float*)   alloc((size_t)NSEQ_ * GNNH_ * 4);
  float*    cnt   = (float*)   alloc((size_t)NSEQ_ * 4);
  _Float16* h1    = (_Float16*)alloc((size_t)NSEQ_ * GNNH_ * 2);
  _Float16* h2    = (_Float16*)alloc((size_t)NSEQ_ * GNNH_ * 2);
  _Float16* q1    = (_Float16*)alloc((size_t)NSEQ_ * 128 * 2);
  _Float16* q2    = (_Float16*)alloc((size_t)NSEQ_ * 128 * 2);
  (void)in_sizes; (void)n_in; (void)out_size; (void)ws_size;

  GArgs a = {};

  soft_k<<<(CITY_ + 191) / 192, 192, 0, stream>>>(wgrp, wsoft);
  uem_k<<<B_, 32, 0, stream>>>(u, emb1, emb2, emb3, ue16);
  lstm_k<<<NSEQ_ / 16, 128, 0, stream>>>(x, Wih, Whh, bih, bhh, xe16);
  gx_k<<<NGRP_, 96, 0, stream>>>(xe16, loc, Wloc, bloc, wsoft, gx);

  // edge inference MLP
  a = {}; a.W = Wedge; a.bias = bedge; a.H0 = gx; a.H1 = ue16; a.OutH = gew; a.K = 184;
  gemm_k<PolEdgeInf><<<NGE_ / 64, 128, 0, stream>>>(a);

  // group GNN layer 1
  hipMemsetAsync(aggG, 0, (size_t)NGRP_ * GNNH_ * 4, stream);
  a = {}; a.W = Wg1m1; a.bias = bg1m1; a.H0 = gx; a.H1 = gew; a.OutF = aggG; a.K = 112;
  gemm_k<PolGMsg1><<<NGE_ / 64, 128, 0, stream>>>(a);
  a = {}; a.W = Wg1m2; a.bias = bg1m2; a.H0 = gx; a.F0 = aggG; a.OutH = gx1; a.K = 144;
  gemm_k<PolGNode1><<<NGRP_ / 64, 128, 0, stream>>>(a);

  // group GNN layer 2
  hipMemsetAsync(aggG, 0, (size_t)NGRP_ * GNNH_ * 4, stream);
  a = {}; a.W = Wg2m1; a.bias = bg2m1; a.H0 = gx1; a.H1 = gew; a.OutF = aggG; a.K = 96;
  gemm_k<PolGMsg2><<<NGE_ / 64, 128, 0, stream>>>(a);
  a = {}; a.W = Wg2m2; a.bias = bg2m2; a.H0 = gx1; a.F0 = aggG; a.OutH = gx2; a.K = 128;
  gemm_k<PolGNode2><<<NGRP_ / 64, 128, 0, stream>>>(a);

  // project back + concat
  newx_k<<<NSEQ_, 64, 0, stream>>>(xe16, gx2, wsoft, nx);

  // global edge in-degree counts (same cols for both layers)
  hipMemsetAsync(cnt, 0, (size_t)NSEQ_ * 4, stream);
  count_k<<<NCE_ / 256, 256, 0, stream>>>(ei, cnt);

  // global GNN layer 1
  hipMemsetAsync(aggC, 0, (size_t)NSEQ_ * GNNH_ * 4, stream);
  a = {}; a.W = Wc1m1; a.bias = bc1m1; a.H0 = nx; a.F0 = ew; a.I0 = ei; a.OutF = aggC; a.K = 129;
  gemm_k<PolCMsg1><<<NCE_ / 64, 128, 0, stream>>>(a);
  a = {}; a.W = Wc1m2; a.bias = bc1m2; a.H0 = nx; a.F0 = aggC; a.F1 = cnt; a.OutH = h1; a.K = 192;
  gemm_k<PolCNode1><<<NSEQ_ / 64, 128, 0, stream>>>(a);

  // global GNN layer 2
  hipMemsetAsync(aggC, 0, (size_t)NSEQ_ * GNNH_ * 4, stream);
  a = {}; a.W = Wc2m1; a.bias = bc2m1; a.H0 = h1; a.F0 = ew; a.I0 = ei; a.OutF = aggC; a.K = 65;
  gemm_k<PolCMsg2><<<NCE_ / 64, 128, 0, stream>>>(a);
  a = {}; a.W = Wc2m2; a.bias = bc2m2; a.H0 = h1; a.F0 = aggC; a.F1 = cnt; a.OutH = h2; a.K = 128;
  gemm_k<PolCNode2><<<NSEQ_ / 64, 128, 0, stream>>>(a);

  // prediction head
  a = {}; a.W = Wp0; a.bias = bp0; a.H0 = h2; a.OutH = q1; a.K = 64;
  gemm_k<PolPred0><<<NSEQ_ / 64, 128, 0, stream>>>(a);
  a = {}; a.W = Wp1; a.bias = bp1; a.H0 = q1; a.OutH = q2; a.K = 128;
  gemm_k<PolPred1><<<NSEQ_ / 64, 128, 0, stream>>>(a);
  pred2_k<<<NSEQ_ / 256, 256, 0, stream>>>(q2, Wp2, bp2, out);
}
